// MultiHeadAttention_30477087932642
// MI455X (gfx1250) — compile-verified, matrix operand order checked by reading, not run on hardware
//
#include <hip/hip_runtime.h>

// ---------------------------------------------------------------------------
// Problem constants (match reference): N=4 batches, n=2048 seq, d=1024.
// ---------------------------------------------------------------------------
constexpr int NB = 4;
constexpr int SEQ = 2048;
constexpr int DIM = 1024;

typedef __bf16 bf16x8  __attribute__((ext_vector_type(8)));
typedef __bf16 bf16x16 __attribute__((ext_vector_type(16)));
typedef float  f32x8   __attribute__((ext_vector_type(8)));

// ---------------------------------------------------------------------------
// CDNA5 async global->LDS copy (ASYNCcnt-tracked). The VDST operand of
// global_load_async_to_lds_b128 is the wave-relative LDS byte address; flat
// addresses of __shared__ objects carry that address in their low 32 bits
// (ISA 10.2).
// ---------------------------------------------------------------------------
#define USE_ASYNC_LDS_COPY 1

__device__ __forceinline__ void async_cp_b128(void* lds_dst, const void* gsrc) {
  const unsigned lds_off = (unsigned)(unsigned long long)lds_dst;
  asm volatile("global_load_async_to_lds_b128 %0, %1, off"
               :: "v"(lds_off), "v"(gsrc) : "memory");
}
__device__ __forceinline__ void wait_async_all() {
  asm volatile("s_wait_asynccnt 0" ::: "memory");
}

// ---------------------------------------------------------------------------
// Fragment loader for V_WMMA_F32_16X16X32_BF16 operands.
// A (16x32) and B-as-rows-over-K (16 rows x 32 K) share the same layout:
//   lane L (0..31): row = L & 15, khalf = L >> 4
//   VGPR 0..3 hold K = khalf*8 .. khalf*8+7   (16 bytes)
//   VGPR 4..7 hold K = khalf*8+16 .. +23      (16 bytes)
// ---------------------------------------------------------------------------
__device__ __forceinline__ bf16x16 load_frag(const __bf16* tile, int ldm, int lane) {
  const int row = lane & 15;
  const int kh  = (lane >> 4) & 1;
  const bf16x8 lo = *(const bf16x8*)(tile + row * ldm + kh * 8);
  const bf16x8 hi = *(const bf16x8*)(tile + row * ldm + kh * 8 + 16);
  return __builtin_shufflevector(lo, hi, 0,1,2,3,4,5,6,7,8,9,10,11,12,13,14,15);
}

// ---------------------------------------------------------------------------
// Generic bf16 GEMM:  C[b][m][j] = epilogue( sum_k A[b][m][k] * B[b][j][k] )
// Both A and B row-major over K ("A * B^T" form) -> identical fragment loads.
// Block tile 128(M) x 256(N) x 32(K); 8 waves as 2(M) x 4(N); each wave owns
// a 64x64 macro-tile: 4 A-frags x 4 B-frags -> 16 WMMAs per 16 ds_load_b128.
// Global->LDS staging is double-buffered with ASYNCcnt-tracked async copies:
// tile i+1 streams into the alternate buffer while tile i feeds the WMMAs.
// MODE 0: store f32. MODE 1: scale + int-mask -> f32. MODE 2: store bf16.
// ---------------------------------------------------------------------------
constexpr int BM = 128, BN = 256, BK = 32;
constexpr int LDSP = 40;  // padded row stride (elements); 80B keeps 16B align,
                          // 20-dword bank advance -> conflict-free frag loads

template <int MODE>
__global__ __launch_bounds__(256) void gemm_bt_kernel(
    const __bf16* __restrict__ A, long long strideA, int lda,
    const __bf16* __restrict__ B, long long strideB, int ldb,
    void* __restrict__ Cout, long long strideC, int ldc,
    const int* __restrict__ mask, long long strideMask,
    int K, float scale) {
  __shared__ __align__(16) __bf16 As[2][BM * LDSP];  // 2 x 10 KB
  __shared__ __align__(16) __bf16 Bs[2][BN * LDSP];  // 2 x 20 KB

  const int b = blockIdx.z;
  A += (long long)b * strideA;
  B += (long long)b * strideB;

  const int tid  = threadIdx.x;
  const int lane = tid & 31;
  const int wave = tid >> 5;   // 0..7
  const int wm   = wave & 1;   // M sub-block (64 rows)
  const int wn   = wave >> 1;  // N sub-block (64 cols)

  const int blockM = blockIdx.y * BM;
  const int blockN = blockIdx.x * BN;

  f32x8 acc[4][4];
  const f32x8 zero = {0.f, 0.f, 0.f, 0.f, 0.f, 0.f, 0.f, 0.f};
#pragma unroll
  for (int i = 0; i < 4; ++i)
#pragma unroll
    for (int j = 0; j < 4; ++j) acc[i][j] = zero;

  // Cooperative staging maps (tiles divide exactly for every GEMM here).
  // A tile 128x32: thread t -> row t/2, 16-elem chunk (t&1)*16  (2 x b128)
  // B tile 256x32: thread t -> row t, full 32-elem row          (4 x b128)
  const int arow = tid >> 1;
  const int acol = (tid & 1) * 16;
  const int brow = tid;

  const __bf16* gA = A + (long long)(blockM + arow) * lda + acol;
  const __bf16* gB = B + (long long)(blockN + brow) * ldb;

  auto issue_tile = [&](int k0, int buf) {
    __bf16* sA = &As[buf][arow * LDSP + acol];
    __bf16* sB = &Bs[buf][brow * LDSP];
#if USE_ASYNC_LDS_COPY
    async_cp_b128(sA,      gA + k0);
    async_cp_b128(sA + 8,  gA + k0 + 8);
    async_cp_b128(sB,      gB + k0);
    async_cp_b128(sB + 8,  gB + k0 + 8);
    async_cp_b128(sB + 16, gB + k0 + 16);
    async_cp_b128(sB + 24, gB + k0 + 24);
#else
    *(bf16x8*)(sA)      = *(const bf16x8*)(gA + k0);
    *(bf16x8*)(sA + 8)  = *(const bf16x8*)(gA + k0 + 8);
    *(bf16x8*)(sB)      = *(const bf16x8*)(gB + k0);
    *(bf16x8*)(sB + 8)  = *(const bf16x8*)(gB + k0 + 8);
    *(bf16x8*)(sB + 16) = *(const bf16x8*)(gB + k0 + 16);
    *(bf16x8*)(sB + 24) = *(const bf16x8*)(gB + k0 + 24);
#endif
  };

  const int nk = K / BK;
  issue_tile(0, 0);

  for (int i = 0; i < nk; ++i) {
    const int cur = i & 1;
#if USE_ASYNC_LDS_COPY
    wait_async_all();  // only buffer `cur`'s 6 copies are in flight here
#endif
    __syncthreads();   // cur visible workgroup-wide; prior reads of nxt done

    if (i + 1 < nk) issue_tile((i + 1) * BK, cur ^ 1);  // overlaps compute

    bf16x16 af[4], bfr[4];
#pragma unroll
    for (int mi = 0; mi < 4; ++mi)
      af[mi] = load_frag(&As[cur][(wm * 64 + mi * 16) * LDSP], LDSP, lane);
#pragma unroll
    for (int nj = 0; nj < 4; ++nj)
      bfr[nj] = load_frag(&Bs[cur][(wn * 64 + nj * 16) * LDSP], LDSP, lane);

#pragma unroll
    for (int mi = 0; mi < 4; ++mi)
#pragma unroll
      for (int nj = 0; nj < 4; ++nj)
        acc[mi][nj] = __builtin_amdgcn_wmma_f32_16x16x32_bf16(
            false, af[mi], false, bfr[nj], (short)0, acc[mi][nj], false, false);

    __syncthreads();   // reads of cur done before cur is refilled at i+2
  }

  // Epilogue. C/D layout: lane L holds column (L&15), rows i + 8*(L>>4).
  const int colL = lane & 15;
  const int rowH = (lane >> 4) * 8;
#pragma unroll
  for (int mi = 0; mi < 4; ++mi) {
#pragma unroll
    for (int ni = 0; ni < 4; ++ni) {
      const int col = blockN + wn * 64 + ni * 16 + colL;
#pragma unroll
      for (int i = 0; i < 8; ++i) {
        const int row = blockM + wm * 64 + mi * 16 + rowH + i;
        const long long cidx =
            (long long)b * strideC + (long long)row * ldc + col;
        float v = acc[mi][ni][i];
        if constexpr (MODE == 0) {
          ((float*)Cout)[cidx] = v;
        } else if constexpr (MODE == 1) {
          const long long midx =
              (long long)b * strideMask + (long long)row * ldc + col;
          v *= scale;
          if (mask[midx] == 0) v = -1e20f;
          ((float*)Cout)[cidx] = v;
        } else {
          ((__bf16*)Cout)[cidx] = (__bf16)v;
        }
      }
    }
  }
}

// ---------------------------------------------------------------------------
// f32 -> bf16 elementwise convert
// ---------------------------------------------------------------------------
__global__ void convert_bf16_kernel(const float* __restrict__ in,
                                    __bf16* __restrict__ out, long long count) {
  long long i = (long long)blockIdx.x * blockDim.x + threadIdx.x;
  long long stride = (long long)gridDim.x * blockDim.x;
  for (; i < count; i += stride) out[i] = (__bf16)in[i];
}

// ---------------------------------------------------------------------------
// Per-batch transpose: Xbf [n x d] -> XT [d x n] (bf16), 32x32 LDS tiles
// ---------------------------------------------------------------------------
__global__ __launch_bounds__(256) void transpose_kernel(
    const __bf16* __restrict__ X, __bf16* __restrict__ XT) {
  __shared__ __bf16 t[32][33];
  const int b  = blockIdx.z;
  const int gx = blockIdx.x * 32;  // d dim
  const int gy = blockIdx.y * 32;  // n dim
  const __bf16* Xb = X + (long long)b * SEQ * DIM;
  __bf16* XTb = XT + (long long)b * DIM * SEQ;
#pragma unroll
  for (int j = 0; j < 4; ++j)
    t[threadIdx.y + j * 8][threadIdx.x] =
        Xb[(long long)(gy + threadIdx.y + j * 8) * DIM + gx + threadIdx.x];
  __syncthreads();
#pragma unroll
  for (int j = 0; j < 4; ++j)
    XTb[(long long)(gx + threadIdx.y + j * 8) * SEQ + gy + threadIdx.x] =
        t[threadIdx.x][threadIdx.y + j * 8];
}

// ---------------------------------------------------------------------------
// Row softmax (f32 in, bf16 out). One 256-thread block per row of length 2048.
// ---------------------------------------------------------------------------
__device__ __forceinline__ float wave_max(float v) {
  for (int off = 16; off; off >>= 1) v = fmaxf(v, __shfl_xor(v, off, 32));
  return v;
}
__device__ __forceinline__ float wave_sum(float v) {
  for (int off = 16; off; off >>= 1) v += __shfl_xor(v, off, 32);
  return v;
}

__global__ __launch_bounds__(256) void softmax_kernel(
    const float* __restrict__ scores, __bf16* __restrict__ P) {
  __shared__ float red[8];
  const long long row = blockIdx.x;
  const float* src = scores + row * SEQ;
  const int tid = threadIdx.x, lane = tid & 31, wave = tid >> 5;

  float v[8];
  float mx = -3.0e38f;
#pragma unroll
  for (int i = 0; i < 8; ++i) {
    v[i] = src[tid + i * 256];
    mx = fmaxf(mx, v[i]);
  }
  mx = wave_max(mx);
  if (lane == 0) red[wave] = mx;
  __syncthreads();
  mx = red[0];
#pragma unroll
  for (int w = 1; w < 8; ++w) mx = fmaxf(mx, red[w]);
  __syncthreads();

  float sum = 0.f;
#pragma unroll
  for (int i = 0; i < 8; ++i) {
    v[i] = __expf(v[i] - mx);
    sum += v[i];
  }
  sum = wave_sum(sum);
  if (lane == 0) red[wave] = sum;
  __syncthreads();
  sum = red[0];
#pragma unroll
  for (int w = 1; w < 8; ++w) sum += red[w];
  const float inv = 1.0f / sum;

  __bf16* dst = P + row * SEQ;
#pragma unroll
  for (int i = 0; i < 8; ++i) dst[tid + i * 256] = (__bf16)(v[i] * inv);
}

// ---------------------------------------------------------------------------
// Host-side orchestration
// ---------------------------------------------------------------------------
extern "C" void kernel_launch(void* const* d_in, const int* in_sizes, int n_in,
                              void* d_out, int out_size, void* d_ws, size_t ws_size,
                              hipStream_t stream) {
  (void)in_sizes; (void)n_in; (void)out_size; (void)ws_size;

  const float* x    = (const float*)d_in[0];  // [N, n, d]
  const int*   mask = (const int*)  d_in[1];  // [N, n, n]
  const float* Wqk  = (const float*)d_in[2];  // [d, d]
  const float* Wvc  = (const float*)d_in[3];  // [d, d]
  float* out = (float*)d_out;                 // [N, n, d]

  const long long XE = (long long)NB * SEQ * DIM;   // 8,388,608
  const long long SE = (long long)NB * SEQ * SEQ;   // 16,777,216
  const long long WE = (long long)DIM * DIM;        // 1,048,576

  // Workspace carving (bytes): ~164 MB total.
  char* ws = (char*)d_ws;
  __bf16* Xbf    = (__bf16*)(ws);                          // 16 MB
  __bf16* XTbf   = (__bf16*)(ws + XE * 2);                 // 16 MB
  __bf16* Qbf    = (__bf16*)(ws + XE * 4);                 // 16 MB
  __bf16* Wqkbf  = (__bf16*)(ws + XE * 6);                 //  2 MB
  __bf16* Wvcbf  = (__bf16*)(ws + XE * 6 + WE * 2);        //  2 MB
  float*  scores = (float*) (ws + XE * 6 + WE * 4);        // 64 MB
  __bf16* Pbf    = (__bf16*)(ws + XE * 6 + WE * 4 + SE * 4);            // 32 MB
  __bf16* attBf  = (__bf16*)(ws + XE * 6 + WE * 4 + SE * 4 + SE * 2);   // 16 MB

  // 1) Precision conversion.
  convert_bf16_kernel<<<2048, 256, 0, stream>>>(x, Xbf, XE);
  convert_bf16_kernel<<<512, 256, 0, stream>>>(Wqk, Wqkbf, WE);
  convert_bf16_kernel<<<512, 256, 0, stream>>>(Wvc, Wvcbf, WE);

  // 2) XT[b] = X[b]^T so the P*X GEMM is also row-major-over-K.
  transpose_kernel<<<dim3(DIM / 32, SEQ / 32, NB), dim3(32, 8), 0, stream>>>(Xbf, XTbf);

  // 3) Q = X * Wqk^T  (M=8192, N=1024, K=1024) -> bf16
  gemm_bt_kernel<2><<<dim3(DIM / BN, (NB * SEQ) / BM, 1), 256, 0, stream>>>(
      Xbf, 0, DIM, Wqkbf, 0, DIM, Qbf, 0, DIM, nullptr, 0, DIM, 1.0f);

  // 4) scores[b] = (Q[b] * X[b]^T) / 32, masked  (M=N=2048, K=1024) -> f32
  gemm_bt_kernel<1><<<dim3(SEQ / BN, SEQ / BM, NB), 256, 0, stream>>>(
      Qbf, (long long)SEQ * DIM, DIM, Xbf, (long long)SEQ * DIM, DIM,
      scores, (long long)SEQ * SEQ, SEQ, mask, (long long)SEQ * SEQ,
      DIM, 0.03125f);

  // 5) P = softmax(scores) rowwise -> bf16
  softmax_kernel<<<NB * SEQ, 256, 0, stream>>>(scores, Pbf);

  // 6) attended[b] = P[b] * XT[b]^T  (M=2048, N=1024, K=2048) -> bf16
  gemm_bt_kernel<2><<<dim3(DIM / BN, SEQ / BM, NB), 256, 0, stream>>>(
      Pbf, (long long)SEQ * SEQ, SEQ, XTbf, (long long)DIM * SEQ, SEQ,
      attBf, (long long)SEQ * DIM, DIM, nullptr, 0, SEQ, 1.0f);

  // 7) out = attended * Wvc^T  (M=8192, N=1024, K=1024) -> f32
  gemm_bt_kernel<0><<<dim3(DIM / BN, (NB * SEQ) / BM, 1), 256, 0, stream>>>(
      attBf, 0, DIM, Wvcbf, 0, DIM, out, 0, DIM, nullptr, 0, DIM, 1.0f);
}
